// Cross_attention_interpro_56813827392375
// MI455X (gfx1250) — matrix operand order repaired and checked by reading, not compiled, not run on hardware
//
#include <hip/hip_runtime.h>
#include <hip/hip_bf16.h>
#include <cstdint>

// ---------------------------------------------------------------------------
// Types
// ---------------------------------------------------------------------------
typedef __bf16 bf16_t;
typedef __bf16 v16bf __attribute__((ext_vector_type(16)));
typedef __bf16 v8bf  __attribute__((ext_vector_type(8)));
typedef float  v8f   __attribute__((ext_vector_type(8)));
typedef unsigned int v4u __attribute__((ext_vector_type(4)));
typedef int v4i_vs __attribute__((vector_size(16)));   // matches async-LDS builtin param

union F16frag { v16bf v; v8bf h[2]; };
union PK8     { v8bf  b; v4u  u; };

#if defined(__HIP_DEVICE_COMPILE__) && defined(__gfx1250__) && \
    __has_builtin(__builtin_amdgcn_global_load_async_to_lds_b128)
#define USE_ASYNC_LDS 1
#else
#define USE_ASYNC_LDS 0
#endif

__device__ __forceinline__ v8f wmma_bf16(const F16frag& a, const F16frag& b, v8f c) {
    return __builtin_amdgcn_wmma_f32_16x16x32_bf16(false, a.v, false, b.v,
                                                   (short)0, c, false, false);
}

__device__ __forceinline__ v8bf cvt8(float4 a, float4 b) {
    v8bf r = {(__bf16)a.x, (__bf16)a.y, (__bf16)a.z, (__bf16)a.w,
              (__bf16)b.x, (__bf16)b.y, (__bf16)b.z, (__bf16)b.w};
    return r;
}

__device__ __forceinline__ v8bf pack8(const float* p) {
    v8bf r = {(__bf16)p[0], (__bf16)p[1], (__bf16)p[2], (__bf16)p[3],
              (__bf16)p[4], (__bf16)p[5], (__bf16)p[6], (__bf16)p[7]};
    return r;
}

// ---------------------------------------------------------------------------
// NT GEMM: Y[m][n] = sum_k A[m][k] * Brows[n][k], K fixed = 1024.
// A, Brows fp32 row-major; Y bf16 (ld = ldy). Tile 128x128, TK=32, 8 waves.
// LDS rows: 32 bf16 (64B) with 16B-chunk XOR swizzle -> conflict-free b128.
// ---------------------------------------------------------------------------
__global__ void __launch_bounds__(256) gemm_nt_bf16(
    const float* __restrict__ A, const float* __restrict__ Bm,
    bf16_t* __restrict__ Y, int lda, int ldb, int ldy) {

    __shared__ bf16_t lA[2][128 * 32];
    __shared__ bf16_t lB[2][128 * 32];

    const int tid = threadIdx.x, lane = tid & 31, wave = tid >> 5;
    const int wm = wave & 1, wn = wave >> 1;          // 2 waves along M, 4 along N
    const int m0 = blockIdx.x * 128, n0 = blockIdx.y * 128;
    const int rr = lane & 15, half = lane >> 4;

    v8f acc[4][2] = {};

    // staging: 512 16B-chunks per matrix, 2 per thread
    size_t aoff[2], boff[2];
    int    doff[2];
#pragma unroll
    for (int j = 0; j < 2; ++j) {
        int cf = tid + j * 256;
        int row = cf >> 2, ch = cf & 3;
        aoff[j] = (size_t)(m0 + row) * lda + ch * 8;
        boff[j] = (size_t)(n0 + row) * ldb + ch * 8;
        doff[j] = row * 32 + ((ch ^ ((row >> 2) & 3)) * 8);
    }

    float4 fa[2][2], fb[2][2];
    auto g_load = [&](int kk) {
#pragma unroll
        for (int j = 0; j < 2; ++j) {
            const float* pa = A + aoff[j] + kk;
            const float* pb = Bm + boff[j] + kk;
            fa[j][0] = *(const float4*)pa;  fa[j][1] = *(const float4*)(pa + 4);
            fb[j][0] = *(const float4*)pb;  fb[j][1] = *(const float4*)(pb + 4);
        }
    };
    auto l_store = [&](int buf) {
#pragma unroll
        for (int j = 0; j < 2; ++j) {
            *(v8bf*)(&lA[buf][doff[j]]) = cvt8(fa[j][0], fa[j][1]);
            *(v8bf*)(&lB[buf][doff[j]]) = cvt8(fb[j][0], fb[j][1]);
        }
    };

    g_load(0);
    l_store(0);
    __syncthreads();

    for (int ki = 0; ki < 32; ++ki) {
        const int cur = ki & 1;
        if (ki < 31) g_load((ki + 1) * 32);   // prefetch next K-slice into regs

        F16frag af[4], bfr[2];
#pragma unroll
        for (int mt = 0; mt < 4; ++mt) {      // A frag: lane=row, K split 0-7/16-23 vs 8-15/24-31
            int row = wm * 64 + mt * 16 + rr;
            int sw = (row >> 2) & 3;
            const bf16_t* bp = &lA[cur][row * 32];
            af[mt].h[0] = *(const v8bf*)(bp + ((half ^ sw) * 8));
            af[mt].h[1] = *(const v8bf*)(bp + (((half + 2) ^ sw) * 8));
        }
#pragma unroll
        for (int nt = 0; nt < 2; ++nt) {      // B frag: lane=col, K split 0-7/16-23 vs 8-15/24-31
            int row = wn * 32 + nt * 16 + rr;
            int sw = (row >> 2) & 3;
            const bf16_t* bp = &lB[cur][row * 32];
            bfr[nt].h[0] = *(const v8bf*)(bp + ((half ^ sw) * 8));
            bfr[nt].h[1] = *(const v8bf*)(bp + (((half + 2) ^ sw) * 8));
        }
#pragma unroll
        for (int mt = 0; mt < 4; ++mt)
#pragma unroll
            for (int nt = 0; nt < 2; ++nt)
                acc[mt][nt] = wmma_bf16(af[mt], bfr[nt], acc[mt][nt]);

        if (ki < 31) l_store(cur ^ 1);
        __syncthreads();
    }

    // epilogue: C layout VGPR i -> M = i + 8*half, N = lane&15
#pragma unroll
    for (int mt = 0; mt < 4; ++mt)
#pragma unroll
        for (int nt = 0; nt < 2; ++nt) {
            int row = m0 + wm * 64 + mt * 16 + half * 8;
            int col = n0 + wn * 32 + nt * 16 + rr;
#pragma unroll
            for (int i = 0; i < 8; ++i)
                Y[(size_t)(row + i) * ldy + col] = (bf16_t)acc[mt][nt][i];
        }
}

// ---------------------------------------------------------------------------
// Flash attention partials. S^T = kh x qh^T (queries on lanes, keys on VGPRs),
// ctx^T = vhT x P. One wave = 16 queries of one head; WG = 8 waves sharing
// K/V LDS tiles; grid = (qhalf=2, chunk=8, head=16); chunk = 4096 keys.
// ---------------------------------------------------------------------------
__global__ void __launch_bounds__(256) attn_partial(
    const bf16_t* __restrict__ qh, const bf16_t* __restrict__ kh,
    const bf16_t* __restrict__ vhT,
    float* __restrict__ pctx, float* __restrict__ pm, float* __restrict__ pl) {

    __shared__ bf16_t lK[2][32 * 64];   // 32 keys x 64 dh
    __shared__ bf16_t lV[2][64 * 32];   // 64 dh x 32 keys (vhT rows)

    const int tid = threadIdx.x, lane = tid & 31, wave = tid >> 5;
    const int rr = lane & 15, half = lane >> 4;
    const int qhalf = blockIdx.x, chunk = blockIdx.y, h = blockIdx.z;
    const int qb = qhalf * 8 + wave;
    const int q0 = qb * 16;
    const int nbase = chunk * 4096;
    const float SC = 0.125f * 1.44269504088896f;   // log2(e)/sqrt(64)

    // qh^T B-fragments (held for whole kernel): lane=query, K=dh
    F16frag bq[2];
    {
        const bf16_t* p = qh + (size_t)(q0 + rr) * 1024 + h * 64 + half * 8;
        bq[0].h[0] = *(const v8bf*)p;        bq[0].h[1] = *(const v8bf*)(p + 16);
        bq[1].h[0] = *(const v8bf*)(p + 32); bq[1].h[1] = *(const v8bf*)(p + 48);
    }

    // staging: 256 threads, one b128 each for K tile (32x64) and V tile (64x32)
    const int krow = tid >> 3, kch = tid & 7;
    const int vrow = tid >> 2, vch = tid & 3;
    const bf16_t* gK0 = kh + (size_t)(nbase + krow) * 1024 + h * 64 + kch * 8;
    const bf16_t* gV0 = vhT + (size_t)(h * 64 + vrow) * 32768 + nbase + vch * 8;
    const int dK = krow * 64 + ((kch ^ ((krow >> 1) & 7)) * 8);
    const int dV = vrow * 32 + ((vch ^ ((vrow >> 2) & 3)) * 8);

    v8bf rk, rv;
    auto stage_issue = [&](int nb, int buf) {
        const bf16_t* gk = gK0 + (size_t)nb * 32 * 1024;
        const bf16_t* gv = gV0 + nb * 32;
        bf16_t* dk = &lK[buf][dK];
        bf16_t* dv = &lV[buf][dV];
#if USE_ASYNC_LDS
        __builtin_amdgcn_global_load_async_to_lds_b128((v4i_vs*)gk, (v4i_vs*)dk, 0, 0);
        __builtin_amdgcn_global_load_async_to_lds_b128((v4i_vs*)gv, (v4i_vs*)dv, 0, 0);
        (void)rk; (void)rv;
#else
        rk = *(const v8bf*)gk;
        rv = *(const v8bf*)gv;
        (void)dk; (void)dv;
#endif
    };
    auto stage_commit = [&](int buf) {
#if !USE_ASYNC_LDS
        *(v8bf*)&lK[buf][dK] = rk;
        *(v8bf*)&lV[buf][dV] = rv;
#else
        (void)buf;
#endif
    };
    auto stage_wait = [&]() {
#if USE_ASYNC_LDS
#if __has_builtin(__builtin_amdgcn_s_wait_asynccnt)
        __builtin_amdgcn_s_wait_asynccnt(0);
#endif
#endif
    };

    v8f co[4] = {};
    float m = -1e30f, l = 0.f;

    stage_issue(0, 0);
    stage_commit(0);
    stage_wait();
    __syncthreads();

    for (int nb = 0; nb < 128; ++nb) {
        const int cur = nb & 1;
        if (nb < 127) stage_issue(nb + 1, cur ^ 1);

        // ---- scores: two 16-key S^T tiles, K over dh=64 (2 wmma each) ----
        v8f s0 = {}, s1 = {};
        {
            F16frag ak;
            int row = rr;                       // keys 0..15
            int sw = (row >> 1) & 7;
            const bf16_t* bp = &lK[cur][row * 64];
            ak.h[0] = *(const v8bf*)(bp + ((half ^ sw) * 8));
            ak.h[1] = *(const v8bf*)(bp + (((half + 2) ^ sw) * 8));
            s0 = wmma_bf16(ak, bq[0], s0);
            ak.h[0] = *(const v8bf*)(bp + (((4 + half) ^ sw) * 8));
            ak.h[1] = *(const v8bf*)(bp + (((6 + half) ^ sw) * 8));
            s0 = wmma_bf16(ak, bq[1], s0);

            row = 16 + rr;                      // keys 16..31
            sw = (row >> 1) & 7;
            bp = &lK[cur][row * 64];
            ak.h[0] = *(const v8bf*)(bp + ((half ^ sw) * 8));
            ak.h[1] = *(const v8bf*)(bp + (((half + 2) ^ sw) * 8));
            s1 = wmma_bf16(ak, bq[0], s1);
            ak.h[0] = *(const v8bf*)(bp + (((4 + half) ^ sw) * 8));
            ak.h[1] = *(const v8bf*)(bp + (((6 + half) ^ sw) * 8));
            s1 = wmma_bf16(ak, bq[1], s1);
        }

        // ---- online softmax (per lane; sync max across lane halves) ----
        float bm = fmaxf(s0[0], s1[0]);
#pragma unroll
        for (int i = 1; i < 8; ++i) bm = fmaxf(bm, fmaxf(s0[i], s1[i]));
        bm = fmaxf(bm, __shfl_xor(bm, 16, 32));
        float mn = fmaxf(m, bm * SC);
        float alpha = __builtin_amdgcn_exp2f(m - mn);

        float p0[8], p1[8], rs = 0.f;
#pragma unroll
        for (int i = 0; i < 8; ++i) { p0[i] = __builtin_amdgcn_exp2f(fmaf(s0[i], SC, -mn)); rs += p0[i]; }
#pragma unroll
        for (int i = 0; i < 8; ++i) { p1[i] = __builtin_amdgcn_exp2f(fmaf(s1[i], SC, -mn)); rs += p1[i]; }
        l = l * alpha + rs;
        m = mn;
#pragma unroll
        for (int t = 0; t < 4; ++t) co[t] = co[t] * alpha;

        // ---- build P B-fragment: per-lane pack only (shuffled o0/o1 unused) ----
        PK8 a0, a1, o0, o1;
        a0.b = pack8(p0);
        a1.b = pack8(p1);
#pragma unroll
        for (int i = 0; i < 4; ++i) {
            o0.u[i] = __shfl_xor(a0.u[i], 16, 32);
            o1.u[i] = __shfl_xor(a1.u[i], 16, 32);
        }
        F16frag pf;
        pf.h[0] = a0.b;   // keys 0..7   | 8..15
        pf.h[1] = a1.b;   // keys 16..23 | 24..31

        // ---- ctx^T += vhT_tile x P ----
#pragma unroll
        for (int t = 0; t < 4; ++t) {
            F16frag av;
            int row = t * 16 + rr;
            int sw = (row >> 2) & 3;
            const bf16_t* bp = &lV[cur][row * 32];
            av.h[0] = *(const v8bf*)(bp + ((half ^ sw) * 8));
            av.h[1] = *(const v8bf*)(bp + (((half + 2) ^ sw) * 8));
            co[t] = wmma_bf16(av, pf, co[t]);
        }

        stage_commit(cur ^ 1);
        stage_wait();
        __syncthreads();
    }

    // ---- write partials ----
    l += __shfl_xor(l, 16, 32);
    const size_t pb = (size_t)((h * 16 + qb) * 8 + chunk);
    float* cd = pctx + pb * 1024;
#pragma unroll
    for (int t = 0; t < 4; ++t)
#pragma unroll
        for (int i = 0; i < 8; ++i)
            cd[(t * 8 + i) * 32 + lane] = co[t][i];
    pm[pb * 32 + lane] = m;
    pl[pb * 32 + lane] = l;
}

// ---------------------------------------------------------------------------
// Combine chunk partials with log-sum-exp and write final fp32 output [B, D].
// ---------------------------------------------------------------------------
__global__ void __launch_bounds__(256) attn_combine(
    const float* __restrict__ pctx, const float* __restrict__ pm,
    const float* __restrict__ pl, float* __restrict__ out) {

    int gid = blockIdx.x * 256 + threadIdx.x;   // 262144 total
    int lane = gid & 31;
    int j = (gid >> 5) & 31;
    int qb = (gid >> 10) & 15;
    int h = gid >> 14;
    int hq = h * 16 + qb;

    float ms[8], M = -1e30f;
#pragma unroll
    for (int c = 0; c < 8; ++c) {
        ms[c] = pm[(size_t)(hq * 8 + c) * 32 + lane];
        M = fmaxf(M, ms[c]);
    }
    float num = 0.f, den = 0.f;
#pragma unroll
    for (int c = 0; c < 8; ++c) {
        float w = __builtin_amdgcn_exp2f(ms[c] - M);
        den += w * pl[(size_t)(hq * 8 + c) * 32 + lane];
        num += w * pctx[(size_t)(hq * 8 + c) * 1024 + j * 32 + lane];
    }
    int q = lane & 15, hf = lane >> 4;
    int dt = j >> 3, i = j & 7;
    int b = qb * 16 + q;
    int dh = dt * 16 + i + hf * 8;
    out[(size_t)b * 1024 + h * 64 + dh] = num / den;
}

// ---------------------------------------------------------------------------
// Launch
// ---------------------------------------------------------------------------
extern "C" void kernel_launch(void* const* d_in, const int* in_sizes, int n_in,
                              void* d_out, int out_size, void* d_ws, size_t ws_size,
                              hipStream_t stream) {
    (void)in_sizes; (void)n_in; (void)out_size; (void)ws_size;

    const float* q  = (const float*)d_in[0];
    const float* k  = (const float*)d_in[1];
    const float* v  = (const float*)d_in[2];
    const float* Wq = (const float*)d_in[3];
    const float* Wk = (const float*)d_in[4];
    const float* Wv = (const float*)d_in[5];

    char* ws = (char*)d_ws;
    bf16_t* qhb  = (bf16_t*)(ws);                    // 512 KiB
    bf16_t* khb  = (bf16_t*)(ws + (1ull  << 20));    // 64 MiB
    bf16_t* vhT  = (bf16_t*)(ws + (65ull << 20));    // 64 MiB
    float*  pctx = (float*)(ws + (129ull << 20));    // 8 MiB
    float*  pm   = (float*)(ws + (138ull << 20));    // 256 KiB
    float*  pl   = (float*)(ws + (139ull << 20));    // 256 KiB

    // qh = q @ Wq^T                 [256 x 1024]
    gemm_nt_bf16<<<dim3(2, 8),   256, 0, stream>>>(q,  Wq, qhb, 1024, 1024, 1024);
    // kh = k @ Wk^T                 [32768 x 1024]
    gemm_nt_bf16<<<dim3(256, 8), 256, 0, stream>>>(k,  Wk, khb, 1024, 1024, 1024);
    // vhT = Wv @ v^T  (pre-transposed V heads)  [1024 x 32768]
    gemm_nt_bf16<<<dim3(8, 256), 256, 0, stream>>>(Wv, v,  vhT, 1024, 1024, 32768);
    // flash-attention partials over 8 N-chunks
    attn_partial<<<dim3(2, 8, 16), 256, 0, stream>>>(qhb, khb, vhT, pctx, pm, pl);
    // LSE combine -> final fp32 output
    attn_combine<<<dim3(1024), 256, 0, stream>>>(pctx, pm, pl, (float*)d_out);
}
